// SimVQ_43937515438691
// MI455X (gfx1250) — matrix operand order repaired and checked by reading, not compiled, not run on hardware
//
#include <hip/hip_runtime.h>
#include <hip/hip_bf16.h>

// ---- types for CDNA5 WMMA / TDM ----
typedef __attribute__((ext_vector_type(16))) __bf16 v16bf;
typedef __attribute__((ext_vector_type(8)))  __bf16 bf16x8;
typedef __attribute__((ext_vector_type(8)))  float  v8f;
typedef unsigned int u32;
typedef __attribute__((ext_vector_type(4))) u32 u32x4;
typedef __attribute__((ext_vector_type(8))) u32 u32x8;

#define D_DIM   512
#define K_CODES 16384
#define N_ROWS  8192          // B*T = 4*2048
#define LDST    520           // padded LDS row stride in bf16 elems (1040B, bank-conflict-free)
#define TROW    1040          // TDM-padded B row stride in bytes (1024B data + 16B pad)
#define TBUF    (16 * TROW)   // one 16-code B tile in LDS = 16640B
#define NWAVE   8

union ABf { v16bf v; bf16x8 h[2]; };

static __device__ __forceinline__ v8f wmma_bf16(v16bf a, v16bf b, v8f c) {
    // D = A(16x32 bf16) * B(32x16 bf16) + C(16x16 f32)
    return __builtin_amdgcn_wmma_f32_16x16x32_bf16(false, a, false, b, (short)0, c, false, false);
}

// ---- Tensor Data Mover: DMA one 16-row x 512-col bf16 tile (16KB) into LDS,
// padding 16B after every 1024B row so the LDS row stride is 1040B. ----
static __device__ __forceinline__ void tdm_load_16x512_bf16(const __bf16* gsrc, u32 lds_off) {
    unsigned long long ga = (unsigned long long)gsrc;
    u32x4 g0;
    g0[0] = 1u;                                              // count=1, user descriptor
    g0[1] = lds_off;                                         // LDS byte address of tile
    g0[2] = (u32)ga;                                         // global_addr[31:0]
    g0[3] = 0x80000000u | ((u32)(ga >> 32) & 0x01FFFFFFu);   // type=2 ("image"), global_addr[56:32]
    u32x8 g1;
    g1[0] = (1u << 16)        // data_size = 2 bytes
          | (1u << 20)        // pad_enable
          | (7u << 22)        // pad_interval = 256 DWORDs (one 1024B row)
          | (3u << 25);       // pad_amount = 4 DWORDs (16B)
    g1[1] = (512u << 16);     // tensor_dim0 = 512  ([63:48] low half)
    g1[2] = (16u  << 16);     // tensor_dim1 = 16   ([95:80] low half)
    g1[3] = (512u << 16);     // tile_dim0  = 512   ([127:112])
    g1[4] = 16u;              // tile_dim1  = 16    ([143:128])
    g1[5] = 512u;             // tensor_dim0_stride = 512 ([191:160])
    g1[6] = 0u;
    g1[7] = 0u;
    asm volatile("tensor_load_to_lds %0, %1" :: "s"(g0), "s"(g1) : "memory");
}

// ---------------- W -> bf16 ----------------
__global__ void k_cvt_w(const float* __restrict__ W, __bf16* __restrict__ Wb, int n) {
    int i = blockIdx.x * blockDim.x + threadIdx.x;
    if (i < n) Wb[i] = (__bf16)W[i];
}

// ------------- eff = codebook @ W.T  (WMMA bf16, f32 accum) -------------
__global__ void __launch_bounds__(256)
k_eff_gemm(const float* __restrict__ cb, const __bf16* __restrict__ Wb,
           float* __restrict__ eff, __bf16* __restrict__ effb) {
    __shared__ __attribute__((aligned(32))) __bf16 sA[16 * LDST];
    const int tid  = threadIdx.x;
    const int row0 = blockIdx.x * 16;
    for (int e = tid; e < 16 * D_DIM; e += 256) {
        int r = e >> 9, d = e & (D_DIM - 1);
        sA[r * LDST + d] = (__bf16)cb[(size_t)(row0 + r) * D_DIM + d];
    }
    __syncthreads();
    const int wave  = tid >> 5, lane = tid & 31;
    const int arow  = lane & 15, col = lane & 15;
    const int abase = (lane < 16) ? 0 : 16;
    const int bbase = (lane < 16) ? 0 : 16;
    for (int t = 0; t < 4; ++t) {
        const int j0 = (wave + 8 * t) * 16;
        v8f acc = {};
#pragma unroll
        for (int c = 0; c < 16; ++c) {
            const int d = c * 32;
            ABf a;
            a.h[0] = *(const bf16x8*)&sA[arow * LDST + d + abase];
            a.h[1] = *(const bf16x8*)&sA[arow * LDST + d + abase + 8];
            const v16bf b = *(const v16bf*)&Wb[(size_t)(j0 + col) * D_DIM + d + bbase];
            acc = wmma_bf16(a.v, b, acc);
        }
#pragma unroll
        for (int i = 0; i < 8; ++i) {
            const int m = i + ((lane < 16) ? 0 : 8);
            const size_t o = (size_t)(row0 + m) * D_DIM + (size_t)(j0 + col);
            eff[o]  = acc[i];
            effb[o] = (__bf16)acc[i];
        }
    }
}

// ---------------- ||eff_k||^2 ----------------
__global__ void __launch_bounds__(128)
k_norms(const float* __restrict__ eff, float* __restrict__ norms) {
    __shared__ float red[128];
    const int k = blockIdx.x, tid = threadIdx.x;
    float s = 0.f;
    for (int d = tid; d < D_DIM; d += 128) { float v = eff[(size_t)k * D_DIM + d]; s += v * v; }
    red[tid] = s; __syncthreads();
    for (int off = 64; off > 0; off >>= 1) { if (tid < off) red[tid] += red[tid + off]; __syncthreads(); }
    if (tid == 0) norms[k] = red[0];
}

// ------- fused distance GEMM + argmin: metric = ||eff||^2 - 2 z.eff -------
// Block owns 32 z-rows (bf16 in LDS). Each of 8 waves owns 2048 codes and
// streams its 16-code B tiles through a TDM double buffer in LDS.
__global__ void __launch_bounds__(256)
k_argmin(const float* __restrict__ z, const __bf16* __restrict__ effb,
         const float* __restrict__ norms, int* __restrict__ idx_out) {
    __shared__ __attribute__((aligned(32))) __bf16 sA[32 * LDST];
    __shared__ float s_val[NWAVE][32];
    __shared__ int   s_idx[NWAVE][32];
    extern __shared__ __attribute__((aligned(32))) char dynlds[];   // NWAVE * 2 * TBUF bytes

    const int tid  = threadIdx.x;
    const int row0 = blockIdx.x * 32;
    for (int e = tid; e < 32 * D_DIM; e += 256) {
        int r = e >> 9, d = e & (D_DIM - 1);
        sA[r * LDST + d] = (__bf16)z[(size_t)(row0 + r) * D_DIM + d];
    }
    __syncthreads();

    const int wave  = tid >> 5, lane = tid & 31;
    const int arow  = lane & 15, col = lane & 15;
    const int abase = (lane < 16) ? 0 : 16;
    const int bbase = (lane < 16) ? 0 : 16;

    // dynamic LDS starts right after (aligned) static LDS
    const u32  dynbase = (__builtin_amdgcn_groupstaticsize() + 31u) & ~31u;
    const u32  wbuf    = dynbase + (u32)wave * 2u * TBUF;            // numeric, for the D#
    char*      wptr    = dynlds + (size_t)wave * 2u * TBUF;          // pointer, for ds reads

    float best0[8], best1[8]; int bi0[8], bi1[8];
#pragma unroll
    for (int i = 0; i < 8; ++i) { best0[i] = best1[i] = 3.4e38f; bi0[i] = bi1[i] = 0; }

    const int kBase = wave * (K_CODES / NWAVE);
    const int NT    = (K_CODES / NWAVE) / 16;                        // 128 tiles per wave

    tdm_load_16x512_bf16(effb + (size_t)kBase * D_DIM, wbuf);        // prime buffer 0
    for (int t = 0; t < NT; ++t) {
        if (t + 1 < NT) {                                            // DMA next tile, overlap compute
            tdm_load_16x512_bf16(effb + (size_t)(kBase + (t + 1) * 16) * D_DIM,
                                 wbuf + (u32)((t + 1) & 1) * TBUF);
            __builtin_amdgcn_s_wait_tensorcnt(1);                    // tile t has landed
        } else {
            __builtin_amdgcn_s_wait_tensorcnt(0);
        }
        const char* bt  = wptr + (size_t)(t & 1) * TBUF;
        const int   tb  = kBase + t * 16;
        const float nrm = norms[tb + col];

        v8f acc0 = {}, acc1 = {};
#pragma unroll
        for (int c = 0; c < 16; ++c) {
            const int d = c * 32;
            ABf a0, a1, b;
            a0.h[0] = *(const bf16x8*)&sA[arow * LDST + d + abase];
            a0.h[1] = *(const bf16x8*)&sA[arow * LDST + d + abase + 8];
            a1.h[0] = *(const bf16x8*)&sA[(arow + 16) * LDST + d + abase];
            a1.h[1] = *(const bf16x8*)&sA[(arow + 16) * LDST + d + abase + 8];
            const bf16x8* bp = (const bf16x8*)(bt + col * TROW + (d + bbase) * 2);
            b.h[0] = bp[0];
            b.h[1] = bp[1];
            acc0 = wmma_bf16(a0.v, b.v, acc0);
            acc1 = wmma_bf16(a1.v, b.v, acc1);
        }
        const int cand = tb + col;
#pragma unroll
        for (int i = 0; i < 8; ++i) {
            float m0 = nrm - 2.f * acc0[i];
            float m1 = nrm - 2.f * acc1[i];
            if (m0 < best0[i]) { best0[i] = m0; bi0[i] = cand; }
            if (m1 < best1[i]) { best1[i] = m1; bi1[i] = cand; }
        }
    }
    // reduce over the 16 columns held within each 16-lane half (xor stays in-half)
#pragma unroll
    for (int m = 1; m < 16; m <<= 1) {
#pragma unroll
        for (int i = 0; i < 8; ++i) {
            float ov = __shfl_xor(best0[i], m, 32); int oi = __shfl_xor(bi0[i], m, 32);
            if (ov < best0[i] || (ov == best0[i] && oi < bi0[i])) { best0[i] = ov; bi0[i] = oi; }
            ov = __shfl_xor(best1[i], m, 32); oi = __shfl_xor(bi1[i], m, 32);
            if (ov < best1[i] || (ov == best1[i] && oi < bi1[i])) { best1[i] = ov; bi1[i] = oi; }
        }
    }
    if (lane == 0 || lane == 16) {
        const int h = (lane == 16) ? 8 : 0;
#pragma unroll
        for (int i = 0; i < 8; ++i) {
            s_val[wave][i + h]      = best0[i]; s_idx[wave][i + h]      = bi0[i];
            s_val[wave][16 + i + h] = best1[i]; s_idx[wave][16 + i + h] = bi1[i];
        }
    }
    __syncthreads();
    if (tid < 32) {
        float bv = s_val[0][tid]; int bi = s_idx[0][tid];
        for (int w = 1; w < NWAVE; ++w) {
            float v = s_val[w][tid]; int ii = s_idx[w][tid];
            if (v < bv || (v == bv && ii < bi)) { bv = v; bi = ii; }
        }
        idx_out[row0 + tid] = bi;
    }
}

// ---------------- gather z_q, loss partial sums, index output ----------------
__global__ void __launch_bounds__(128)
k_gather(const float* __restrict__ z, const float* __restrict__ eff,
         const int* __restrict__ idx, float* __restrict__ zq_out,
         float* __restrict__ idxf, float* __restrict__ acc) {
    __shared__ float red[128];
    const int n = blockIdx.x, tid = threadIdx.x;
    const int k = idx[n];
    float s = 0.f;
    for (int d = tid; d < D_DIM; d += 128) {
        float q = eff[(size_t)k * D_DIM + d];
        float e = z[(size_t)n * D_DIM + d];
        zq_out[(size_t)n * D_DIM + d] = q;        // straight-through forward == z_q
        float df = e - q; s += df * df;
    }
    red[tid] = s; __syncthreads();
    for (int off = 64; off > 0; off >>= 1) { if (tid < off) red[tid] += red[tid + off]; __syncthreads(); }
    if (tid == 0) { atomicAdd(acc, red[0]); idxf[n] = (float)k; }
}

__global__ void k_zero(float* acc) { if (threadIdx.x == 0 && blockIdx.x == 0) *acc = 0.f; }

__global__ void k_final(const float* __restrict__ acc, float* __restrict__ loss) {
    // total = mse + 0.25*mse  (stop_gradient doesn't change forward value)
    *loss = 1.25f * (*acc) / (float)((size_t)N_ROWS * D_DIM);
}

extern "C" void kernel_launch(void* const* d_in, const int* in_sizes, int n_in,
                              void* d_out, int out_size, void* d_ws, size_t ws_size,
                              hipStream_t stream) {
    const float* z  = (const float*)d_in[0];   // (4,2048,512)
    const float* cb = (const float*)d_in[1];   // (16384,512)
    const float* W  = (const float*)d_in[2];   // (512,512)

    char* ws = (char*)d_ws;
    float*  eff   = (float*)(ws);                                             // 33.55 MB
    __bf16* effb  = (__bf16*)(ws + (size_t)K_CODES * D_DIM * 4);              // 16.78 MB
    __bf16* Wb    = (__bf16*)(ws + (size_t)K_CODES * D_DIM * 6);              // 0.52 MB
    float*  norms = (float*)(ws + (size_t)K_CODES * D_DIM * 6
                                + (size_t)D_DIM * D_DIM * 2);                 // 64 KB
    int*    idx   = (int*)((char*)norms + (size_t)K_CODES * 4);               // 32 KB
    float*  acc   = (float*)((char*)idx + (size_t)N_ROWS * 4);                // 4 B

    float* zq_out = (float*)d_out;                         // N*D
    float* loss   = zq_out + (size_t)N_ROWS * D_DIM;       // 1
    float* idxf   = loss + 1;                              // N (indices as float)

    const size_t dyn_lds = (size_t)NWAVE * 2 * TBUF;       // 260 KB TDM double buffers

    k_cvt_w   <<<(D_DIM * D_DIM + 255) / 256, 256, 0, stream>>>(W, Wb, D_DIM * D_DIM);
    k_eff_gemm<<<K_CODES / 16, 256, 0, stream>>>(cb, Wb, eff, effb);
    k_norms   <<<K_CODES, 128, 0, stream>>>(eff, norms);
    k_zero    <<<1, 32, 0, stream>>>(acc);
    k_argmin  <<<N_ROWS / 32, 256, dyn_lds, stream>>>(z, effb, norms, idx);
    k_gather  <<<N_ROWS, 128, 0, stream>>>(z, eff, idx, zq_out, idxf, acc);
    k_final   <<<1, 1, 0, stream>>>(acc, loss);
}
